// EncLayer_74637941670410
// MI455X (gfx1250) — matrix operand order repaired and checked by reading, not compiled, not run on hardware
//
#include <hip/hip_runtime.h>

#define Hdim 128
#define Kn   32
#define DINd 384
#define FFd  512
#define SCALE_ 30.0f
#define EPSF 1e-5f

typedef __attribute__((ext_vector_type(4)))  unsigned short u16x4;
typedef __attribute__((ext_vector_type(8)))  unsigned short u16x8;
typedef __attribute__((ext_vector_type(16))) unsigned short u16x16;
typedef __attribute__((ext_vector_type(16))) __bf16         bf16x16;
typedef __attribute__((ext_vector_type(8)))  float          f32x8;

// ---------- small helpers ----------
__device__ __forceinline__ unsigned short f2b(float f){
  unsigned u = __builtin_bit_cast(unsigned, f);
  u += 0x7FFFu + ((u >> 16) & 1u);            // round-to-nearest-even bf16
  return (unsigned short)(u >> 16);
}
__device__ __forceinline__ u16x4 f2b4(float4 v){
  u16x4 r; r[0] = f2b(v.x); r[1] = f2b(v.y); r[2] = f2b(v.z); r[3] = f2b(v.w);
  return r;
}
__device__ __forceinline__ float gelu_erf(float x){
  return 0.5f * x * (1.0f + erff(x * 0.70710678118654752f));
}
__device__ __forceinline__ float wave_sum(float v){
#pragma unroll
  for (int off = 16; off >= 1; off >>= 1) v += __shfl_xor(v, off, 32);
  return v;
}
// async 16B global->LDS copy (CDNA5 path, tracked by ASYNCcnt)
__device__ __forceinline__ void async_cp16(unsigned lds_byte_addr, const void* gptr){
  asm volatile("global_load_async_to_lds_b128 %0, %1, off"
               :: "v"(lds_byte_addr), "v"(gptr) : "memory");
}
__device__ __forceinline__ void async_wait0(){
  asm volatile("s_wait_asynccnt 0x0" ::: "memory");
}

// A fragment: lane holds row (lane&15), K chunks [k+8h, +8) and [k+16+8h, +8)  (ISA 7.12.2)
__device__ __forceinline__ bf16x16 ldA16(const unsigned short* p){
  u16x8 lo = *reinterpret_cast<const u16x8*>(p);
  u16x8 hi = *reinterpret_cast<const u16x8*>(p + 16);
  u16x16 t = __builtin_shufflevector(lo, hi, 0,1,2,3,4,5,6,7,8,9,10,11,12,13,14,15);
  return __builtin_bit_cast(bf16x16, t);
}
// B fragment: lane holds column (lane&15), K chunks [k+8h, +8) and [k+16+8h, +8)
__device__ __forceinline__ bf16x16 ldB16(const unsigned short* p){
  u16x8 lo = *reinterpret_cast<const u16x8*>(p);
  u16x8 hi = *reinterpret_cast<const u16x8*>(p + 16);
  u16x16 t = __builtin_shufflevector(lo, hi, 0,1,2,3,4,5,6,7,8,9,10,11,12,13,14,15);
  return __builtin_bit_cast(bf16x16, t);
}

// 32xKD (LDS, bf16, row stride AS) @ KDx16 column tile of Bt (bf16, [out][in]) -> two 16x16 f32 frags
template<int KD, int AS>
__device__ __forceinline__ void gemm_tile(const unsigned short* A, const unsigned short* Bt,
                                          int ncol0, int lane, f32x8& c0, f32x8& c1){
  const int half = lane >> 4;
  const int l15  = lane & 15;
  const unsigned short* a0p = A + l15 * AS + (half << 3);
  const unsigned short* a1p = a0p + 16 * AS;
  const unsigned short* bp  = Bt + (size_t)(ncol0 + l15) * KD + (half << 3);
#pragma unroll
  for (int k = 0; k < KD; k += 32){
    bf16x16 a0 = ldA16(a0p + k);
    bf16x16 a1 = ldA16(a1p + k);
    bf16x16 b  = ldB16(bp + k);
    c0 = __builtin_amdgcn_wmma_f32_16x16x32_bf16(false, a0, false, b, (short)0, c0, false, false);
    c1 = __builtin_amdgcn_wmma_f32_16x16x32_bf16(false, a1, false, b, (short)0, c1, false, false);
  }
}

// D frag element i lives at (M = mbase + 8*(lane>>4) + i, N = col)
__device__ __forceinline__ void epi_gelu_store(const f32x8& c, int mbase, int lane, int col,
                                               float bias, unsigned short* dst, int ds){
  int mo = mbase + ((lane >> 4) << 3);
#pragma unroll
  for (int i = 0; i < 8; ++i)
    dst[(mo + i) * ds + col] = f2b(gelu_erf(c[i] + bias));
}

// LayerNorm over a 128-float LDS row, 1 wave, 4 features/lane
__device__ __forceinline__ void row_ln4(const float* row, int lane,
                                        const float* g, const float* b, float y[4]){
  float a[4];
#pragma unroll
  for (int j = 0; j < 4; ++j) a[j] = row[4*lane + j];
  float mean = wave_sum(a[0] + a[1] + a[2] + a[3]) * (1.0f/Hdim);
  float d[4], sq = 0.f;
#pragma unroll
  for (int j = 0; j < 4; ++j){ d[j] = a[j] - mean; sq += d[j]*d[j]; }
  float var = wave_sum(sq) * (1.0f/Hdim);
  float r = rsqrtf(var + EPSF);
#pragma unroll
  for (int j = 0; j < 4; ++j) y[j] = d[j]*r*g[4*lane + j] + b[4*lane + j];
}

// ---------- weight prep: fp32 [in][out] -> bf16 [out][in] ----------
__global__ void transpose_bf16(const float* __restrict__ src, unsigned short* __restrict__ dst,
                               int in_dim, int out_dim){
  int i = blockIdx.x * blockDim.x + threadIdx.x;
  if (i >= in_dim * out_dim) return;
  int r = i / out_dim, c = i - r * out_dim;
  dst[(size_t)c * in_dim + r] = f2b(src[i]);
}

// ---------- kernel A: node message MLP + masked sum + LN1 ----------
__global__ __launch_bounds__(256) void node_msg_kernel(
    const float* __restrict__ hV, const float* __restrict__ hE,
    const float* __restrict__ maskA, const int* __restrict__ E_idx,
    const unsigned short* __restrict__ W1t, const float* __restrict__ b1,
    const unsigned short* __restrict__ W2t, const float* __restrict__ b2,
    const unsigned short* __restrict__ W3t, const float* __restrict__ b3,
    const float* __restrict__ ln1g, const float* __restrict__ ln1b,
    float* __restrict__ hv1f, unsigned short* __restrict__ hv1b)
{
  const int node = blockIdx.x;
  const int tid = threadIdx.x, lane = tid & 31, w = tid >> 5;

  __shared__ unsigned short sEV[Kn][DINd + 8];
  __shared__ unsigned short sM1[Kn][Hdim + 8];
  __shared__ unsigned short sM2[Kn][Hdim + 8];
  __shared__ float sMask[Kn];
  __shared__ int   sIdx[Kn];
  __shared__ float sDh[Hdim];
  __shared__ float sRa[8], sRb[8];

  if (tid < Kn){
    int nb = E_idx[node*Kn + tid];
    sIdx[tid]  = nb;
    sMask[tid] = maskA[node*Kn + tid];
    __builtin_prefetch(hV + (size_t)nb * Hdim, 0, 0);   // global_prefetch for gather
  }
  __syncthreads();

  // stage h_EV (f32 -> bf16, float4-vectorized): 32 rows x (128|128|128)
  for (int i = tid; i < (Kn*Hdim)/4; i += 256){
    int r = i >> 5, c4 = (i & 31) << 2;
    float4 vc = *(const float4*)&hV[(size_t)node*Hdim + c4];
    float4 ve = *(const float4*)&hE[((size_t)node*Kn + r)*Hdim + c4];
    float4 vn = *(const float4*)&hV[(size_t)sIdx[r]*Hdim + c4];
    *(u16x4*)&sEV[r][c4]          = f2b4(vc);
    *(u16x4*)&sEV[r][Hdim + c4]   = f2b4(ve);
    *(u16x4*)&sEV[r][2*Hdim + c4] = f2b4(vn);
  }
  __syncthreads();

  const int nc = w * 16, col = nc + (lane & 15);
  {
    f32x8 c0 = {}, c1 = {};
    gemm_tile<DINd, DINd+8>(&sEV[0][0], W1t, nc, lane, c0, c1);
    float bb = b1[col];
    epi_gelu_store(c0, 0,  lane, col, bb, &sM1[0][0], Hdim+8);
    epi_gelu_store(c1, 16, lane, col, bb, &sM1[0][0], Hdim+8);
  }
  __syncthreads();
  {
    f32x8 c0 = {}, c1 = {};
    gemm_tile<Hdim, Hdim+8>(&sM1[0][0], W2t, nc, lane, c0, c1);
    float bb = b2[col];
    epi_gelu_store(c0, 0,  lane, col, bb, &sM2[0][0], Hdim+8);
    epi_gelu_store(c1, 16, lane, col, bb, &sM2[0][0], Hdim+8);
  }
  __syncthreads();
  {
    f32x8 c0 = {}, c1 = {};
    gemm_tile<Hdim, Hdim+8>(&sM2[0][0], W3t, nc, lane, c0, c1);
    float bb = b3[col];
    int mo = (lane >> 4) << 3;
    float part = 0.f;
#pragma unroll
    for (int i = 0; i < 8; ++i){
      part += (c0[i] + bb) * sMask[mo + i];
      part += (c1[i] + bb) * sMask[16 + mo + i];
    }
    part += __shfl_xor(part, 16, 32);   // combine half-waves (same column)
    if (lane < 16) sDh[col] = part;
  }
  __syncthreads();

  // fused LN1 over 128 features (threads 0..127 hold one feature each)
  float x = 0.f;
  if (tid < Hdim) x = hV[(size_t)node*Hdim + tid] + sDh[tid] * (1.0f/SCALE_);
  float s = wave_sum(x);
  if (lane == 0) sRa[w] = s;
  __syncthreads();
  float mean = (sRa[0]+sRa[1]+sRa[2]+sRa[3]) * (1.0f/Hdim);
  float d = x - mean;
  float sq = wave_sum((tid < Hdim) ? d*d : 0.f);
  if (lane == 0) sRb[w] = sq;
  __syncthreads();
  float var = (sRb[0]+sRb[1]+sRb[2]+sRb[3]) * (1.0f/Hdim);
  if (tid < Hdim){
    float y = d * rsqrtf(var + EPSF) * ln1g[tid] + ln1b[tid];
    hv1f[(size_t)node*Hdim + tid] = y;
    hv1b[(size_t)node*Hdim + tid] = f2b(y);
  }
}

// ---------- kernel B: FFN (32 nodes / block) + LN2 + mask_V ----------
__global__ __launch_bounds__(256) void ffn_kernel(
    const float* __restrict__ hv1f, const unsigned short* __restrict__ hv1b,
    const unsigned short* __restrict__ Wint, const float* __restrict__ binw,
    const unsigned short* __restrict__ Woutt, const float* __restrict__ bout,
    const float* __restrict__ ln2g, const float* __restrict__ ln2b,
    const float* __restrict__ maskV,
    float* __restrict__ outV, unsigned short* __restrict__ hvb, int Ntot)
{
  const int node0 = blockIdx.x * 32;
  const int tid = threadIdx.x, lane = tid & 31, w = tid >> 5;
  __shared__ unsigned short sX[32][Hdim + 8];
  __shared__ unsigned short sF[32][FFd + 8];
  __shared__ float sY[32][Hdim];

  // async copy 32x128 bf16 tile into LDS (16B chunks; global stride 256B, LDS stride 272B)
  {
    unsigned base = (unsigned)(size_t)(void*)&sX[0][0];
    for (int q = tid; q < 32*(Hdim/8); q += 256){  // 512 chunks
      int r = q >> 4, ce = (q & 15) << 3;
      int n = node0 + r; if (n >= Ntot) n = Ntot - 1;
      async_cp16(base + (unsigned)((r*(Hdim+8) + ce)*2),
                 hv1b + (size_t)n*Hdim + ce);
    }
    async_wait0();
  }
  __syncthreads();
#pragma unroll
  for (int t = 0; t < 4; ++t){
    int nc = (w + t*8) * 16, col = nc + (lane & 15);
    f32x8 c0 = {}, c1 = {};
    gemm_tile<Hdim, Hdim+8>(&sX[0][0], Wint, nc, lane, c0, c1);
    float bb = binw[col];
    epi_gelu_store(c0, 0,  lane, col, bb, &sF[0][0], FFd+8);
    epi_gelu_store(c1, 16, lane, col, bb, &sF[0][0], FFd+8);
  }
  __syncthreads();
  {
    int nc = w*16, col = nc + (lane & 15);
    f32x8 c0 = {}, c1 = {};
    gemm_tile<FFd, FFd+8>(&sF[0][0], Woutt, nc, lane, c0, c1);
    float bb = bout[col];
    int mo = (lane >> 4) << 3;
#pragma unroll
    for (int i = 0; i < 8; ++i){
      int n0 = node0 + mo + i;      if (n0 >= Ntot) n0 = Ntot - 1;
      int n1 = node0 + 16 + mo + i; if (n1 >= Ntot) n1 = Ntot - 1;
      sY[mo+i][col]    = c0[i] + bb + hv1f[(size_t)n0*Hdim + col];
      sY[16+mo+i][col] = c1[i] + bb + hv1f[(size_t)n1*Hdim + col];
    }
  }
  __syncthreads();
#pragma unroll
  for (int t = 0; t < 4; ++t){
    int r = w + t*8, n = node0 + r;
    float y[4];
    row_ln4(sY[r], lane, ln2g, ln2b, y);
    if (n < Ntot){
      float mk = maskV[n];
#pragma unroll
      for (int j = 0; j < 4; ++j){
        float v = y[j] * mk;
        outV[(size_t)n*Hdim + 4*lane + j] = v;
        hvb [(size_t)n*Hdim + 4*lane + j] = f2b(v);
      }
    }
  }
}

// ---------- kernel C: edge message MLP + residual + LN3 ----------
__global__ __launch_bounds__(256) void edge_kernel(
    const unsigned short* __restrict__ hvb, const float* __restrict__ hE,
    const int* __restrict__ E_idx,
    const unsigned short* __restrict__ W11t, const float* __restrict__ b11,
    const unsigned short* __restrict__ W12t, const float* __restrict__ b12,
    const unsigned short* __restrict__ W13t, const float* __restrict__ b13,
    const float* __restrict__ ln3g, const float* __restrict__ ln3b,
    float* __restrict__ outE)
{
  const int node = blockIdx.x;
  const int tid = threadIdx.x, lane = tid & 31, w = tid >> 5;
  __shared__ unsigned short sEV[Kn][DINd + 8];
  __shared__ unsigned short sM1[Kn][Hdim + 8];
  __shared__ unsigned short sM2[Kn][Hdim + 8];
  __shared__ float sXY[Kn][Hdim];
  __shared__ int sIdx[Kn];

  if (tid < Kn){
    int nb = E_idx[node*Kn + tid];
    sIdx[tid] = nb;
    __builtin_prefetch(hvb + (size_t)nb*Hdim, 0, 0);
  }
  __syncthreads();

  // center + neighbor thirds are already bf16 in global -> async copy straight to LDS.
  {
    unsigned base = (unsigned)(size_t)(void*)&sEV[0][0];
    for (int q = tid; q < 2*Kn*(Hdim/8); q += 256){   // 1024 chunks of 16B
      int sec = q >> 9;                                // 0: center, 1: neighbor
      int r   = (q >> 4) & 31;
      int ce  = (q & 15) << 3;
      int srcnode = sec ? sIdx[r] : node;
      int dcol    = sec ? (2*Hdim + ce) : ce;
      async_cp16(base + (unsigned)((r*(DINd+8) + dcol)*2),
                 hvb + (size_t)srcnode*Hdim + ce);
    }
  }
  // middle third: h_E f32 -> bf16 on the VALU while async copies fly
  for (int i = tid; i < (Kn*Hdim)/4; i += 256){
    int r = i >> 5, c4 = (i & 31) << 2;
    float4 ve = *(const float4*)&hE[((size_t)node*Kn + r)*Hdim + c4];
    *(u16x4*)&sEV[r][Hdim + c4] = f2b4(ve);
  }
  async_wait0();
  __syncthreads();

  const int nc = w*16, col = nc + (lane & 15);
  {
    f32x8 c0 = {}, c1 = {};
    gemm_tile<DINd, DINd+8>(&sEV[0][0], W11t, nc, lane, c0, c1);
    float bb = b11[col];
    epi_gelu_store(c0, 0,  lane, col, bb, &sM1[0][0], Hdim+8);
    epi_gelu_store(c1, 16, lane, col, bb, &sM1[0][0], Hdim+8);
  }
  __syncthreads();
  {
    f32x8 c0 = {}, c1 = {};
    gemm_tile<Hdim, Hdim+8>(&sM1[0][0], W12t, nc, lane, c0, c1);
    float bb = b12[col];
    epi_gelu_store(c0, 0,  lane, col, bb, &sM2[0][0], Hdim+8);
    epi_gelu_store(c1, 16, lane, col, bb, &sM2[0][0], Hdim+8);
  }
  __syncthreads();
  {
    f32x8 c0 = {}, c1 = {};
    gemm_tile<Hdim, Hdim+8>(&sM2[0][0], W13t, nc, lane, c0, c1);
    float bb = b13[col];
    int mo = (lane >> 4) << 3;
#pragma unroll
    for (int i = 0; i < 8; ++i){
      sXY[mo+i][col]    = c0[i] + bb + hE[((size_t)node*Kn + mo+i)*Hdim + col];
      sXY[16+mo+i][col] = c1[i] + bb + hE[((size_t)node*Kn + 16+mo+i)*Hdim + col];
    }
  }
  __syncthreads();
#pragma unroll
  for (int t = 0; t < 4; ++t){
    int r = w + t*8;
    float y[4];
    row_ln4(sXY[r], lane, ln3g, ln3b, y);
#pragma unroll
    for (int j = 0; j < 4; ++j)
      outE[((size_t)node*Kn + r)*Hdim + 4*lane + j] = y[j];
  }
}

// ---------- host launch ----------
extern "C" void kernel_launch(void* const* d_in, const int* in_sizes, int n_in,
                              void* d_out, int out_size, void* d_ws, size_t ws_size,
                              hipStream_t stream)
{
  (void)n_in; (void)out_size; (void)ws_size;
  const float* hV    = (const float*)d_in[0];
  const float* hE    = (const float*)d_in[1];
  const float* maskV = (const float*)d_in[2];
  const float* maskA = (const float*)d_in[3];
  const float* W1  = (const float*)d_in[4];  const float* b1  = (const float*)d_in[5];
  const float* W2  = (const float*)d_in[6];  const float* b2  = (const float*)d_in[7];
  const float* W3  = (const float*)d_in[8];  const float* b3  = (const float*)d_in[9];
  const float* W11 = (const float*)d_in[10]; const float* b11 = (const float*)d_in[11];
  const float* W12 = (const float*)d_in[12]; const float* b12 = (const float*)d_in[13];
  const float* W13 = (const float*)d_in[14]; const float* b13 = (const float*)d_in[15];
  const float* Win = (const float*)d_in[16]; const float* binw= (const float*)d_in[17];
  const float* Wout= (const float*)d_in[18]; const float* bout= (const float*)d_in[19];
  const float* ln1g= (const float*)d_in[20]; const float* ln1b= (const float*)d_in[21];
  const float* ln2g= (const float*)d_in[22]; const float* ln2b= (const float*)d_in[23];
  const float* ln3g= (const float*)d_in[24]; const float* ln3b= (const float*)d_in[25];
  const int* E_idx = (const int*)d_in[26];

  const int N = in_sizes[0] / Hdim;

  char* ws = (char*)d_ws;
  size_t off = 0;
  auto alloc = [&](size_t bytes) -> void* {
    void* p = ws + off;
    off = (off + bytes + 255) & ~(size_t)255;
    return p;
  };
  unsigned short* W1t   = (unsigned short*)alloc((size_t)128*384*2);
  unsigned short* W2t   = (unsigned short*)alloc((size_t)128*128*2);
  unsigned short* W3t   = (unsigned short*)alloc((size_t)128*128*2);
  unsigned short* W11t  = (unsigned short*)alloc((size_t)128*384*2);
  unsigned short* W12t  = (unsigned short*)alloc((size_t)128*128*2);
  unsigned short* W13t  = (unsigned short*)alloc((size_t)128*128*2);
  unsigned short* Wint  = (unsigned short*)alloc((size_t)512*128*2);
  unsigned short* Woutt = (unsigned short*)alloc((size_t)128*512*2);
  float*          hv1f  = (float*)         alloc((size_t)N*Hdim*4);
  unsigned short* hv1b  = (unsigned short*)alloc((size_t)N*Hdim*2);
  unsigned short* hvb   = (unsigned short*)alloc((size_t)N*Hdim*2);

  auto tp = [&](const float* s, unsigned short* d, int in_dim, int out_dim){
    int total = in_dim * out_dim;
    transpose_bf16<<<(total + 255)/256, 256, 0, stream>>>(s, d, in_dim, out_dim);
  };
  tp(W1,  W1t,  DINd, Hdim);  tp(W2,  W2t,  Hdim, Hdim);  tp(W3,  W3t,  Hdim, Hdim);
  tp(W11, W11t, DINd, Hdim);  tp(W12, W12t, Hdim, Hdim);  tp(W13, W13t, Hdim, Hdim);
  tp(Win, Wint, Hdim, FFd);   tp(Wout, Woutt, FFd, Hdim);

  float* outV = (float*)d_out;
  float* outE = outV + (size_t)N * Hdim;

  node_msg_kernel<<<N, 256, 0, stream>>>(hV, hE, maskA, E_idx,
                                         W1t, b1, W2t, b2, W3t, b3,
                                         ln1g, ln1b, hv1f, hv1b);
  ffn_kernel<<<(N + 31)/32, 256, 0, stream>>>(hv1f, hv1b, Wint, binw, Woutt, bout,
                                              ln2g, ln2b, maskV, outV, hvb, N);
  edge_kernel<<<N, 256, 0, stream>>>(hvb, hE, E_idx,
                                     W11t, b11, W12t, b12, W13t, b13,
                                     ln3g, ln3b, outE);
}
